// HiggsAudioTransformer_34943853920433
// MI455X (gfx1250) — matrix operand order repaired and checked by reading, not compiled, not run on hardware
//
#include <hip/hip_runtime.h>
#include <hip/hip_bf16.h>

typedef __attribute__((ext_vector_type(16))) __bf16 bf16x16;
typedef __attribute__((ext_vector_type(8)))  __bf16 bf16x8;
typedef __attribute__((ext_vector_type(8)))  float  f32x8;
typedef __attribute__((ext_vector_type(4)))  int    i32x4;

#define TOKENS 2048      // B*S
#define HID    2048
#define QKVD   4096
#define FFND   8192
#define SEQ    1024
#define HDIM   128
#define NHEAD  16
#define NKVH   8
#define NBATCH 32        // B*NHEAD
#define TEXTV  32000
#define SLICE  1026
#define ATTN_SCALE 0.08838834764831845f

// --- CDNA5 async global->LDS copy support (probe at compile time) ----------
#if defined(__has_builtin)
#  if __has_builtin(__builtin_amdgcn_global_load_async_to_lds_b128) && \
      __has_builtin(__builtin_amdgcn_s_wait_asynccnt)
#    define ASYNC_COPY 1
#  endif
#endif
#ifndef ASYNC_COPY
#  define ASYNC_COPY 0
#endif

#if ASYNC_COPY
typedef __attribute__((address_space(1))) i32x4 gas_i32x4;  // global
typedef __attribute__((address_space(3))) i32x4 las_i32x4;  // LDS
#define GLD16(ldsdst, gsrc)                                              \
  __builtin_amdgcn_global_load_async_to_lds_b128(                        \
      (gas_i32x4*)(void*)(gsrc), (las_i32x4*)(void*)(ldsdst), 0, 0)
#define NBUF 2
#else
#define NBUF 1
#endif

// ---------------------------------------------------------------------------
// Embedding: text path or sum of 8 codebook slices; also writes audio mask.
// ---------------------------------------------------------------------------
__global__ __launch_bounds__(256) void embed_kernel(
    const int* __restrict__ ids, const float* __restrict__ ve,
    const float* __restrict__ ae, float* __restrict__ h, int* __restrict__ mask) {
  int tok = blockIdx.x;
  int id = ids[tok];
  int isa = (id > TEXTV - 1) ? 1 : 0;
  if (threadIdx.x == 0) mask[tok] = isa;
  long ho = (long)tok * HID;
  if (!isa) {
    const float* src = ve + (long)id * HID;
    for (int j = threadIdx.x; j < HID; j += 256) h[ho + j] = src[j];
  } else {
    int aid = id - TEXTV;
    for (int j = threadIdx.x; j < HID; j += 256) {
      float acc = 0.f;
#pragma unroll
      for (int c = 0; c < 8; ++c) acc += ae[(long)(aid + c * SLICE) * HID + j];
      h[ho + j] = acc;
    }
  }
}

// ---------------------------------------------------------------------------
// Transpose + convert: W[K][N] fp32 -> Wt[N][K] bf16 (so B operand is K-contig)
// ---------------------------------------------------------------------------
__global__ __launch_bounds__(256) void transpose_convert(
    const float* __restrict__ W, __bf16* __restrict__ Wt, int K, int N) {
  __shared__ float t[32][33];
  int n0 = blockIdx.x * 32, k0 = blockIdx.y * 32;
  int tx = threadIdx.x, ty = threadIdx.y; // 32 x 8
#pragma unroll
  for (int i = 0; i < 4; ++i)
    t[ty * 4 + i][tx] = W[(long)(k0 + ty * 4 + i) * N + n0 + tx];
  __syncthreads();
#pragma unroll
  for (int i = 0; i < 4; ++i)
    Wt[(long)(n0 + ty * 4 + i) * K + k0 + tx] = (__bf16)t[tx][ty * 4 + i];
}

// ---------------------------------------------------------------------------
// Dual-path RMSNorm. mode 0: single bf16 out, weight chosen by mask.
//                    mode 1: two bf16 outs (text / audio weights).
// ---------------------------------------------------------------------------
__global__ __launch_bounds__(256) void rmsnorm_dual(
    const float* __restrict__ h, const float* __restrict__ w0,
    const float* __restrict__ w1, const int* __restrict__ mask,
    __bf16* __restrict__ out0, __bf16* __restrict__ out1, int mode) {
  __shared__ float red[256];
  int tok = blockIdx.x;
  const float* x = h + (long)tok * HID;
  float ss = 0.f;
  for (int j = threadIdx.x; j < HID; j += 256) { float v = x[j]; ss += v * v; }
  red[threadIdx.x] = ss; __syncthreads();
  for (int s = 128; s > 0; s >>= 1) {
    if (threadIdx.x < s) red[threadIdx.x] += red[threadIdx.x + s];
    __syncthreads();
  }
  float r = rsqrtf(red[0] * (1.0f / HID) + 1e-5f);
  long o = (long)tok * HID;
  if (mode == 0) {
    const float* w = mask[tok] ? w1 : w0;
    for (int j = threadIdx.x; j < HID; j += 256)
      out0[o + j] = (__bf16)(x[j] * r * w[j]);
  } else {
    for (int j = threadIdx.x; j < HID; j += 256) {
      float v = x[j] * r;
      out0[o + j] = (__bf16)(v * w0[j]);
      out1[o + j] = (__bf16)(v * w1[j]);
    }
  }
}

__global__ __launch_bounds__(256) void final_rms(
    const float* __restrict__ h, const float* __restrict__ w, float* __restrict__ out) {
  __shared__ float red[256];
  int tok = blockIdx.x;
  const float* x = h + (long)tok * HID;
  float ss = 0.f;
  for (int j = threadIdx.x; j < HID; j += 256) { float v = x[j]; ss += v * v; }
  red[threadIdx.x] = ss; __syncthreads();
  for (int s = 128; s > 0; s >>= 1) {
    if (threadIdx.x < s) red[threadIdx.x] += red[threadIdx.x + s];
    __syncthreads();
  }
  float r = rsqrtf(red[0] * (1.0f / HID) + 1e-5f);
  long o = (long)tok * HID;
  for (int j = threadIdx.x; j < HID; j += 256) out[o + j] = x[j] * r * w[j];
}

// ---------------------------------------------------------------------------
// Pack qkv (fp32) -> head-major bf16 Q[b*h][s][d], K[b*h][s][d] (GQA-replicated),
// Vt[b*h][d][s] (transposed, GQA-replicated).
// ---------------------------------------------------------------------------
__global__ __launch_bounds__(256) void pack_qkv(
    const float* __restrict__ qkv, __bf16* __restrict__ Qb,
    __bf16* __restrict__ Kb, __bf16* __restrict__ Vtb) {
  int tok = blockIdx.x;
  int b = tok >> 10, s = tok & 1023;
  const float* row = qkv + (long)tok * QKVD;
  for (int i = threadIdx.x; i < NHEAD * HDIM; i += 256) {
    int hh = i >> 7, d = i & 127;
    int z = b * NHEAD + hh;
    int kv = hh >> 1;
    Qb[((long)z * SEQ + s) * HDIM + d]  = (__bf16)row[hh * HDIM + d];
    Kb[((long)z * SEQ + s) * HDIM + d]  = (__bf16)row[NHEAD * HDIM + kv * HDIM + d];
    Vtb[((long)z * HDIM + d) * SEQ + s] = (__bf16)row[(NHEAD + NKVH) * HDIM + kv * HDIM + d];
  }
}

// ---------------------------------------------------------------------------
// Causal softmax: scores fp32 [z][m][n] -> probs bf16, scale folded here.
// ---------------------------------------------------------------------------
__global__ __launch_bounds__(256) void softmax_causal(
    const float* __restrict__ scores, __bf16* __restrict__ probs) {
  __shared__ float red[256];
  int m = blockIdx.x, z = blockIdx.y;
  const float* srow = scores + ((long)z * SEQ + m) * SEQ;
  __bf16* prow = probs + ((long)z * SEQ + m) * SEQ;
  float sv[4];
  float mx = -3.4e38f;
#pragma unroll
  for (int j = 0; j < 4; ++j) {
    int n = threadIdx.x + j * 256;
    sv[j] = (n <= m) ? srow[n] * ATTN_SCALE : -3.4e38f;
    mx = fmaxf(mx, sv[j]);
  }
  red[threadIdx.x] = mx; __syncthreads();
  for (int s = 128; s > 0; s >>= 1) {
    if (threadIdx.x < s) red[threadIdx.x] = fmaxf(red[threadIdx.x], red[threadIdx.x + s]);
    __syncthreads();
  }
  mx = red[0]; __syncthreads();
  float sum = 0.f;
#pragma unroll
  for (int j = 0; j < 4; ++j) {
    int n = threadIdx.x + j * 256;
    sv[j] = (n <= m) ? __expf(sv[j] - mx) : 0.f;
    sum += sv[j];
  }
  red[threadIdx.x] = sum; __syncthreads();
  for (int s = 128; s > 0; s >>= 1) {
    if (threadIdx.x < s) red[threadIdx.x] += red[threadIdx.x + s];
    __syncthreads();
  }
  float inv = 1.f / red[0];
#pragma unroll
  for (int j = 0; j < 4; ++j) {
    int n = threadIdx.x + j * 256;
    prow[n] = (__bf16)(sv[j] * inv);
  }
}

__global__ __launch_bounds__(256) void f32_to_bf16(
    const float* __restrict__ in, __bf16* __restrict__ out, long n) {
  long i = (long)blockIdx.x * 256 + threadIdx.x;
  if (i < n) out[i] = (__bf16)in[i];
}

// ---------------------------------------------------------------------------
// Generic batched bf16 WMMA GEMM: C = A[MxK] * Bt[NxK]^T
// Block tile 128x128, K step 32. 256 threads = 8 waves (2 M x 4 N),
// each wave: 64x32 via 4x2 grid of v_wmma_f32_16x16x32_bf16.
// Double-buffered async global->LDS pipeline (ASYNCcnt) when available.
// Epilogues: 0 store f32, 1 silu->bf16, 2 add to f32, 3 masked add to f32.
// C batch offset = (z>>czshift)*cs1 + (z&((1<<czshift)-1))*cs2.
// ---------------------------------------------------------------------------
#define LDSP 40  // padded row length (bf16 elems) for 32-wide K tiles

template <int EPI>
__global__ __launch_bounds__(256) void gemm_bf16(
    const __bf16* __restrict__ A, long sA, int lda,
    const __bf16* __restrict__ B, long sB, int ldb,
    float* __restrict__ Cf, __bf16* __restrict__ Cb, int ldc,
    int czshift, long cs1, long cs2, int K,
    const int* __restrict__ mask, int maskSel) {
  __shared__ __align__(16) __bf16 sAt[NBUF][128 * LDSP];
  __shared__ __align__(16) __bf16 sBt[NBUF][128 * LDSP];

  const int tid = threadIdx.x;
  const int wave = tid >> 5, lane = tid & 31;
  const int wm = wave >> 2, wn = wave & 3;
  const int z = blockIdx.z;

  const __bf16* Ab = A + (long)z * sA + (long)(blockIdx.y * 128) * lda;
  const __bf16* Bb = B + (long)z * sB + (long)(blockIdx.x * 128) * ldb;
  const long coff = (long)(z >> czshift) * cs1 +
                    (long)(z & ((1 << czshift) - 1)) * cs2;

  f32x8 acc[4][2] = {};

  const int lrow = lane & 15;
  const int akb = (lane >> 4) * 16;  // A half-wave K base (0 or 16)
  const int bkb = (lane >> 4) * 16;  // B half-wave K base (0 or 16)

  // per-thread cooperative-load coordinates: 256 threads x 16 bf16 = 128x32
  const int r = tid >> 1, hh = (tid & 1) * 16;
  const __bf16* agp = Ab + (long)r * lda + hh;
  const __bf16* bgp = Bb + (long)r * ldb + hh;
  const int sOff = r * LDSP + hh;
  const int nk = K >> 5;

#if ASYNC_COPY
  // prologue: fill buffer 0 for k-tile 0
  GLD16(&sAt[0][sOff], agp);
  GLD16(&sAt[0][sOff + 8], agp + 8);
  GLD16(&sBt[0][sOff], bgp);
  GLD16(&sBt[0][sOff + 8], bgp + 8);
#endif

  for (int ki = 0; ki < nk; ++ki) {
#if ASYNC_COPY
    const int cur = ki & 1;
    __builtin_amdgcn_s_wait_asynccnt(0);  // current buffer complete
    __syncthreads();                       // visible to all; prev reads done
    if (ki + 1 < nk) {                     // prefetch next tile (overlaps WMMA)
      const int nxt = cur ^ 1;
      const __bf16* ag = agp + ((ki + 1) << 5);
      const __bf16* bg = bgp + ((ki + 1) << 5);
      GLD16(&sAt[nxt][sOff], ag);
      GLD16(&sAt[nxt][sOff + 8], ag + 8);
      GLD16(&sBt[nxt][sOff], bg);
      GLD16(&sBt[nxt][sOff + 8], bg + 8);
    }
#else
    const int cur = 0;
    {
      const __bf16* ap = agp + (ki << 5);
      bf16x8 a0 = *(const bf16x8*)ap;
      bf16x8 a1 = *(const bf16x8*)(ap + 8);
      *(bf16x8*)&sAt[0][sOff] = a0;
      *(bf16x8*)&sAt[0][sOff + 8] = a1;
      const __bf16* bp = bgp + (ki << 5);
      bf16x8 b0 = *(const bf16x8*)bp;
      bf16x8 b1 = *(const bf16x8*)(bp + 8);
      *(bf16x8*)&sBt[0][sOff] = b0;
      *(bf16x8*)&sBt[0][sOff + 8] = b1;
    }
    __syncthreads();
#endif

    bf16x16 af[4];
#pragma unroll
    for (int rr = 0; rr < 4; ++rr) {
      const __bf16* p = &sAt[cur][(wm * 64 + rr * 16 + lrow) * LDSP + akb];
      bf16x8 lo = *(const bf16x8*)p;        // K = akb..akb+7
      bf16x8 hi = *(const bf16x8*)(p + 8);  // K = akb+8..akb+15
      af[rr] = __builtin_shufflevector(lo, hi, 0, 1, 2, 3, 4, 5, 6, 7,
                                       8, 9, 10, 11, 12, 13, 14, 15);
    }
    bf16x16 bfr[2];
#pragma unroll
    for (int c = 0; c < 2; ++c) {
      const __bf16* p = &sBt[cur][(wn * 32 + c * 16 + lrow) * LDSP + bkb];
      bf16x8 lo = *(const bf16x8*)p;       // K = bkb..bkb+7
      bf16x8 hi = *(const bf16x8*)(p + 8); // K = bkb+8..bkb+15
      bfr[c] = __builtin_shufflevector(lo, hi, 0, 1, 2, 3, 4, 5, 6, 7,
                                       8, 9, 10, 11, 12, 13, 14, 15);
    }
#pragma unroll
    for (int rr = 0; rr < 4; ++rr)
#pragma unroll
      for (int c = 0; c < 2; ++c)
        acc[rr][c] = __builtin_amdgcn_wmma_f32_16x16x32_bf16(
            false, af[rr], false, bfr[c], (short)0, acc[rr][c], false, false);
#if !ASYNC_COPY
    __syncthreads();
#endif
  }

  // epilogue: VGPR i, lanes 0-15 -> M=i, lanes 16-31 -> M=8+i; N = lane&15
  const int rofs = (lane >> 4) * 8;
#pragma unroll
  for (int rr = 0; rr < 4; ++rr)
#pragma unroll
    for (int c = 0; c < 2; ++c)
#pragma unroll
      for (int i = 0; i < 8; ++i) {
        int gm = blockIdx.y * 128 + wm * 64 + rr * 16 + rofs + i;
        int gn = blockIdx.x * 128 + wn * 32 + c * 16 + lrow;
        long idx = coff + (long)gm * ldc + gn;
        float v = acc[rr][c][i];
        if (EPI == 0) {
          Cf[idx] = v;
        } else if (EPI == 1) {
          Cb[idx] = (__bf16)(v / (1.f + __expf(-v)));
        } else if (EPI == 2) {
          Cf[idx] += v;
        } else {
          if (mask[gm] == maskSel) Cf[idx] += v;
        }
      }
}

// ---------------------------------------------------------------------------
extern "C" void kernel_launch(void* const* d_in, const int* in_sizes, int n_in,
                              void* d_out, int out_size, void* d_ws, size_t ws_size,
                              hipStream_t stream) {
  const int*   input_ids   = (const int*)d_in[0];
  const float* vocab_embed = (const float*)d_in[1];
  const float* audio_embed = (const float*)d_in[2];
  const float* ln_in       = (const float*)d_in[3];
  const float* ln_audio_in = (const float*)d_in[4];
  const float* ln_post     = (const float*)d_in[5];
  const float* ln_audio_post = (const float*)d_in[6];
  const float* w_qkv  = (const float*)d_in[7];
  const float* w_o    = (const float*)d_in[8];
  const float* w_fc   = (const float*)d_in[9];
  const float* w_proj = (const float*)d_in[10];
  const float* w_afc  = (const float*)d_in[11];
  const float* w_aproj = (const float*)d_in[12];
  const float* ln_f   = (const float*)d_in[13];
  float* out = (float*)d_out;

  char* ws = (char*)d_ws;
  size_t off = 0;
  auto alloc = [&](size_t bytes) -> void* {
    void* p = ws + off;
    off += (bytes + 255) & ~(size_t)255;
    return p;
  };

  float*  h      = (float*)alloc((size_t)TOKENS * HID * 4);
  int*    mask   = (int*)alloc((size_t)TOKENS * 4);
  __bf16* xbf    = (__bf16*)alloc((size_t)TOKENS * HID * 2);
  __bf16* xabf   = (__bf16*)alloc((size_t)TOKENS * HID * 2);
  float*  qkvf   = (float*)alloc((size_t)TOKENS * QKVD * 4);
  __bf16* Qb     = (__bf16*)alloc((size_t)NBATCH * SEQ * HDIM * 2);
  __bf16* Kb     = (__bf16*)alloc((size_t)NBATCH * SEQ * HDIM * 2);
  __bf16* Vtb    = (__bf16*)alloc((size_t)NBATCH * HDIM * SEQ * 2);
  float*  scores = (float*)alloc((size_t)NBATCH * SEQ * SEQ * 4);
  __bf16* probs  = (__bf16*)alloc((size_t)NBATCH * SEQ * SEQ * 2);
  float*  Of     = (float*)alloc((size_t)TOKENS * HID * 4);
  __bf16* Obf    = (__bf16*)alloc((size_t)TOKENS * HID * 2);
  __bf16* ffn1   = (__bf16*)alloc((size_t)TOKENS * FFND * 2);
  __bf16* wqkvT  = (__bf16*)alloc((size_t)QKVD * HID * 2);
  __bf16* woT    = (__bf16*)alloc((size_t)HID * HID * 2);
  __bf16* wfcT   = (__bf16*)alloc((size_t)FFND * HID * 2);
  __bf16* wprojT = (__bf16*)alloc((size_t)HID * FFND * 2);
  __bf16* wafcT  = (__bf16*)alloc((size_t)FFND * HID * 2);
  __bf16* waprojT = (__bf16*)alloc((size_t)HID * FFND * 2);

  embed_kernel<<<TOKENS, 256, 0, stream>>>(input_ids, vocab_embed, audio_embed, h, mask);

  const dim3 tb(32, 8);
  for (int l = 0; l < 2; ++l) {
    const float* wqkv_l  = w_qkv  + (size_t)l * HID * QKVD;
    const float* wo_l    = w_o    + (size_t)l * HID * HID;
    const float* wfc_l   = w_fc   + (size_t)l * HID * FFND;
    const float* wproj_l = w_proj + (size_t)l * FFND * HID;
    const float* wafc_l  = w_afc  + (size_t)l * HID * FFND;
    const float* waproj_l = w_aproj + (size_t)l * FFND * HID;

    transpose_convert<<<dim3(QKVD / 32, HID / 32), tb, 0, stream>>>(wqkv_l, wqkvT, HID, QKVD);
    transpose_convert<<<dim3(HID / 32, HID / 32), tb, 0, stream>>>(wo_l, woT, HID, HID);
    transpose_convert<<<dim3(FFND / 32, HID / 32), tb, 0, stream>>>(wfc_l, wfcT, HID, FFND);
    transpose_convert<<<dim3(HID / 32, FFND / 32), tb, 0, stream>>>(wproj_l, wprojT, FFND, HID);
    transpose_convert<<<dim3(FFND / 32, HID / 32), tb, 0, stream>>>(wafc_l, wafcT, HID, FFND);
    transpose_convert<<<dim3(HID / 32, FFND / 32), tb, 0, stream>>>(waproj_l, waprojT, FFND, HID);

    // input norm (mask-merged) -> qkv GEMM
    rmsnorm_dual<<<TOKENS, 256, 0, stream>>>(h, ln_in + l * HID, ln_audio_in + l * HID,
                                             mask, xbf, nullptr, 0);
    gemm_bf16<0><<<dim3(QKVD / 128, TOKENS / 128, 1), 256, 0, stream>>>(
        xbf, 0, HID, wqkvT, 0, HID, qkvf, nullptr, QKVD, 0, 0, 0, HID, nullptr, 0);

    pack_qkv<<<TOKENS, 256, 0, stream>>>(qkvf, Qb, Kb, Vtb);

    // scores = Q K^T  (batched over B*NH)
    gemm_bf16<0><<<dim3(SEQ / 128, SEQ / 128, NBATCH), 256, 0, stream>>>(
        Qb, (long)SEQ * HDIM, HDIM, Kb, (long)SEQ * HDIM, HDIM,
        scores, nullptr, SEQ, 0, (long)SEQ * SEQ, 0, HDIM, nullptr, 0);

    softmax_causal<<<dim3(SEQ, NBATCH), 256, 0, stream>>>(scores, probs);

    // O = P V  (C written head-merged: [b][q][h*128+d]); z>>4 = batch b, z&15 = head
    gemm_bf16<0><<<dim3(HDIM / 128, SEQ / 128, NBATCH), 256, 0, stream>>>(
        probs, (long)SEQ * SEQ, SEQ, Vtb, (long)HDIM * SEQ, SEQ,
        Of, nullptr, HID, 4, (long)SEQ * HID, HDIM, SEQ, nullptr, 0);

    f32_to_bf16<<<((long)TOKENS * HID + 255) / 256, 256, 0, stream>>>(Of, Obf, (long)TOKENS * HID);

    // h += O @ w_o
    gemm_bf16<2><<<dim3(HID / 128, TOKENS / 128, 1), 256, 0, stream>>>(
        Obf, 0, HID, woT, 0, HID, h, nullptr, HID, 0, 0, 0, HID, nullptr, 0);

    // dual-path FFN
    rmsnorm_dual<<<TOKENS, 256, 0, stream>>>(h, ln_post + l * HID, ln_audio_post + l * HID,
                                             mask, xbf, xabf, 1);
    // text path: silu(x @ wfc) @ wproj, added where mask==0
    gemm_bf16<1><<<dim3(FFND / 128, TOKENS / 128, 1), 256, 0, stream>>>(
        xbf, 0, HID, wfcT, 0, HID, nullptr, ffn1, FFND, 0, 0, 0, HID, nullptr, 0);
    gemm_bf16<3><<<dim3(HID / 128, TOKENS / 128, 1), 256, 0, stream>>>(
        ffn1, 0, FFND, wprojT, 0, FFND, h, nullptr, HID, 0, 0, 0, FFND, mask, 0);
    // audio path: added where mask==1
    gemm_bf16<1><<<dim3(FFND / 128, TOKENS / 128, 1), 256, 0, stream>>>(
        xabf, 0, HID, wafcT, 0, HID, nullptr, ffn1, FFND, 0, 0, 0, HID, nullptr, 0);
    gemm_bf16<3><<<dim3(HID / 128, TOKENS / 128, 1), 256, 0, stream>>>(
        ffn1, 0, FFND, waprojT, 0, FFND, h, nullptr, HID, 0, 0, 0, FFND, mask, 1);
  }

  final_rms<<<TOKENS, 256, 0, stream>>>(h, ln_f, out);
}
